// Model_24541443130057
// MI455X (gfx1250) — matrix operand order repaired and checked by reading, not compiled, not run on hardware
//
#include <hip/hip_runtime.h>
#include <math.h>

typedef __attribute__((ext_vector_type(4)))  _Float16 v4h;
typedef __attribute__((ext_vector_type(8)))  _Float16 v8h;
typedef __attribute__((ext_vector_type(16))) _Float16 v16h;
typedef __attribute__((ext_vector_type(4)))  float    v4f;
typedef __attribute__((ext_vector_type(8)))  float    v8f;

#define NB 8
#define NV 321
#define NL 512
#define NPRED 96
#define ND 128
#define NP 64
#define NBV (NB*NV)          // 2568
#define NT ((size_t)NBV*NP)  // 164352

// ---------------- per-(b,v) mean / std over L ----------------
__global__ void stats_kernel(const float* __restrict__ x_enc, float* mean, float* stdv) {
  int bv = blockIdx.x;
  int b = bv / NV, v = bv % NV;
  int tid = threadIdx.x; // 128
  __shared__ float s1[128], s2[128];
  float a = 0.f, sq = 0.f;
  for (int l = tid; l < NL; l += 128) {
    float x = x_enc[((size_t)b*NL + l)*NV + v];
    a += x; sq += x*x;
  }
  s1[tid] = a; s2[tid] = sq; __syncthreads();
  for (int s = 64; s > 0; s >>= 1) {
    if (tid < s) { s1[tid] += s1[tid+s]; s2[tid] += s2[tid+s]; }
    __syncthreads();
  }
  if (tid == 0) {
    float m = s1[0] / (float)NL;
    float var = s2[0]/(float)NL - m*m;
    mean[bv] = m;
    stdv[bv] = sqrtf(var + 1e-5f);
  }
}

// ---------------- patch embed + positional encoding ----------------
__global__ void embed_kernel(const float* __restrict__ x_enc, const float* mean, const float* stdv,
                             const float* __restrict__ W_emb, float* __restrict__ out) {
  int bv = blockIdx.x, p = blockIdx.y;
  int b = bv / NV, v = bv % NV;
  int d = threadIdx.x; // 128
  __shared__ float patch[16];
  if (d < 16) {
    int i = p*8 + d;
    int src = (i < NL) ? i : (NL-1);          // right-pad repeats last value
    patch[d] = (x_enc[((size_t)b*NL + src)*NV + v] - mean[bv]) / stdv[bv];
  }
  __syncthreads();
  int i2 = d >> 1;
  float div = expf((float)(2*i2) * (-9.2103403719761836f/128.f));  // -ln(1e4)/D
  float ang = (float)p * div;
  float acc = (d & 1) ? cosf(ang) : sinf(ang);
  #pragma unroll
  for (int k = 0; k < 16; k++) acc += patch[k] * W_emb[k*ND + d];
  out[((size_t)bv*NP + p)*ND + d] = acc;
}

// ---------------- generic WMMA GEMM: C = A(f32) @ B(f32) [+C] [+bias] [gelu] ----------------
// block = 256 threads (8 waves in 4x2), block tile 128x64, per-wave tile 32x32, k-step 32.
// Requirements (all call sites satisfy): K % 32 == 0, N % 4 == 0, ldb % 4 == 0,
// lda % 4 == 0, 16-byte aligned base pointers.
__global__ void gemm_wmma(const float* __restrict__ A, const float* __restrict__ Bw,
                          const float* __restrict__ bias, float* __restrict__ C,
                          int M, int N, int K, int lda, int ldb, int ldc,
                          int act, int beta)
{
  __shared__ _Float16 As[128][40];  // [m][k], row padded to 40 halfs (80B, 16B-aligned)
  __shared__ _Float16 Bs[64][40];   // [n][k]  (transposed tile), padded
  int tid  = threadIdx.x;
  int wave = tid >> 5;        // 0..7
  int wm   = wave >> 1;       // 0..3 -> M sub-tile (32 rows)
  int wn   = wave & 1;        // 0..1 -> N sub-tile (32 cols)
  int lane = tid & 31;
  int half = lane >> 4;       // 0/1
  int l16  = lane & 15;
  int m0 = blockIdx.x * 128;
  int n0 = blockIdx.y * 64;

  // A staging map: thread -> row ar, 16 contiguous k at ac (128x32 tile, 256 threads)
  int ar = tid >> 1;
  int ac = (tid & 1) * 16;
  // B staging map (first 128 threads): 4x4 sub-blocks: rows br..br+3 (k), cols bc..bc+3 (n)
  int br = ((tid & 127) >> 4) * 4;
  int bc = (tid & 15) * 4;

  v8f acc[2][2];
  #pragma unroll
  for (int i = 0; i < 2; i++)
    #pragma unroll
    for (int j = 0; j < 2; j++) acc[i][j] = (v8f){0,0,0,0,0,0,0,0};

  for (int k0 = 0; k0 < K; k0 += 32) {
    // ---- stage A tile (128 x 32) as f16, wide loads/stores ----
    {
      int gm = m0 + ar;
      float t[16];
      if (gm < M) {
        const float* ap = A + (size_t)gm*lda + k0 + ac;
        #pragma unroll
        for (int j = 0; j < 4; j++) {
          v4f q = *(const v4f*)(ap + 4*j);
          t[4*j+0] = q[0]; t[4*j+1] = q[1]; t[4*j+2] = q[2]; t[4*j+3] = q[3];
        }
        if (k0 + 32 < K) __builtin_prefetch(ap + 32, 0, 3);  // keep near (WGP scope)
      } else {
        #pragma unroll
        for (int j = 0; j < 16; j++) t[j] = 0.f;
      }
      v8h lo, hi;
      #pragma unroll
      for (int e = 0; e < 8; e++) { lo[e] = (_Float16)t[e]; hi[e] = (_Float16)t[8+e]; }
      *(v8h*)&As[ar][ac]     = lo;
      *(v8h*)&As[ar][ac + 8] = hi;
    }
    // ---- stage B tile (32 x 64) transposed -> Bs[n][k] (threads 0..127) ----
    if (tid < 128) {
      float t[4][4];
      int gn = n0 + bc;
      if (gn < N) {  // N % 4 == 0 -> whole 4-col group in or out
        const float* bp = Bw + (size_t)(k0 + br)*ldb + gn;
        #pragma unroll
        for (int r = 0; r < 4; r++) {
          v4f q = *(const v4f*)(bp + (size_t)r*ldb);
          t[r][0] = q[0]; t[r][1] = q[1]; t[r][2] = q[2]; t[r][3] = q[3];
        }
        if (k0 + 32 < K) __builtin_prefetch(bp + (size_t)32*ldb, 0, 3);
      } else {
        #pragma unroll
        for (int r = 0; r < 4; r++)
          #pragma unroll
          for (int c = 0; c < 4; c++) t[r][c] = 0.f;
      }
      #pragma unroll
      for (int c = 0; c < 4; c++) {
        v4h col;
        #pragma unroll
        for (int r = 0; r < 4; r++) col[r] = (_Float16)t[r][c];
        *(v4h*)&Bs[bc + c][br] = col;
      }
    }
    __syncthreads();

    // ---- fragments: 2 A-frags + 2 B-frags -> 4 WMMA ----
    v16h afrag[2], bfrag[2];
    #pragma unroll
    for (int mi = 0; mi < 2; mi++) {
      // A: lanes 0-15 K{0..7,16..23}, lanes 16-31 K{8..15,24..31}
      int arow = wm*32 + mi*16 + l16;
      v8h alo = *(const v8h*)&As[arow][half*8];
      v8h ahi = *(const v8h*)&As[arow][16 + half*8];
      afrag[mi] = __builtin_shufflevector(alo, ahi,
          0,1,2,3,4,5,6,7,8,9,10,11,12,13,14,15);
    }
    #pragma unroll
    for (int ni = 0; ni < 2; ni++) {
      // B: lane column, lanes 0-15 K{0..7,16..23}, lanes 16-31 K{8..15,24..31}
      int col = wn*32 + ni*16 + l16;
      v8h blo = *(const v8h*)&Bs[col][half*8];
      v8h bhi = *(const v8h*)&Bs[col][16 + half*8];
      bfrag[ni] = __builtin_shufflevector(blo, bhi,
          0,1,2,3,4,5,6,7,8,9,10,11,12,13,14,15);
    }
    #pragma unroll
    for (int mi = 0; mi < 2; mi++)
      #pragma unroll
      for (int ni = 0; ni < 2; ni++)
        acc[mi][ni] = __builtin_amdgcn_wmma_f32_16x16x32_f16(
            false, afrag[mi], false, bfrag[ni], (short)0, acc[mi][ni], false, false);
    __syncthreads();
  }

  // epilogue: C/D layout -> row = r + 8*half, col = l16
  #pragma unroll
  for (int mi = 0; mi < 2; mi++) {
    #pragma unroll
    for (int ni = 0; ni < 2; ni++) {
      #pragma unroll
      for (int r = 0; r < 8; r++) {
        int row = m0 + wm*32 + mi*16 + half*8 + r;
        int col = n0 + wn*32 + ni*16 + l16;
        if (row < M && col < N) {
          float v = acc[mi][ni][r];
          if (beta) v += C[(size_t)row*ldc + col];
          if (bias) v += bias[col];
          if (act == 1) { // gelu (tanh approx, jax default)
            float x = v, x3 = x*x*x;
            v = 0.5f*x*(1.f + tanhf(0.7978845608f*(x + 0.044715f*x3)));
          }
          C[(size_t)row*ldc + col] = v;
        }
      }
    }
  }
}

// ---------------- depthwise conv (D_CONV=2) + silu; rev flips causality ----------------
__global__ void conv_silu_kernel(const float* __restrict__ xz, const float* __restrict__ w,
                                 const float* __restrict__ b, float* __restrict__ xc, int rev) {
  size_t idx = (size_t)blockIdx.x*256 + threadIdx.x;
  if (idx >= NT*ND) return;
  int d = (int)(idx & (ND-1));
  size_t t = idx >> 7;
  int p = (int)(t & (NP-1));
  float cur = xz[t*(2*ND) + d];
  float other = 0.f;
  if (!rev) { if (p > 0)    other = xz[(t-1)*(2*ND) + d]; }
  else      { if (p < NP-1) other = xz[(t+1)*(2*ND) + d]; }
  float v = cur*w[d*2+1] + other*w[d*2+0] + b[d];
  xc[idx] = v / (1.f + expf(-v));  // silu
}

// ---------------- selective scan: one block per (b,v) sequence, thread = channel d ----------------
__global__ void scan_kernel(const float* __restrict__ dbc, const float* __restrict__ xc,
                            const float* __restrict__ W_dt, const float* __restrict__ b_dt,
                            const float* __restrict__ A_log, const float* __restrict__ Dvec,
                            float* __restrict__ ys, int rev) {
  int bv = blockIdx.x;
  int d = threadIdx.x; // 128
  __shared__ float sh[40]; // [0:8) dt-proj input, [8:24) B, [24:40) C
  float wdt[8];
  #pragma unroll
  for (int r = 0; r < 8; r++) wdt[r] = W_dt[r*ND + d];
  float bdt = b_dt[d];
  float Av[16];
  #pragma unroll
  for (int n = 0; n < 16; n++) Av[n] = -expf(A_log[d*16 + n]);
  float Dp = Dvec[d];
  float h[16];
  #pragma unroll
  for (int n = 0; n < 16; n++) h[n] = 0.f;

  for (int s = 0; s < NP; s++) {
    int p = rev ? (NP-1-s) : s;
    size_t t = (size_t)bv*NP + p;
    if (d < 40) sh[d] = dbc[t*40 + d];
    __syncthreads();
    float xcv = xc[t*ND + d];
    float a = bdt;
    #pragma unroll
    for (int r = 0; r < 8; r++) a += sh[r]*wdt[r];
    float dt = (a > 20.f) ? a : log1pf(expf(a));   // softplus
    float y = 0.f;
    #pragma unroll
    for (int n = 0; n < 16; n++) {
      float dA = expf(dt * Av[n]);
      h[n] = dA*h[n] + dt*sh[8+n]*xcv;
      y += h[n]*sh[24+n];
    }
    ys[t*ND + d] = y + xcv*Dp;
    __syncthreads();
  }
}

// ---------------- y *= silu(z) (z = second half of xz) ----------------
__global__ void mul_silu_kernel(float* __restrict__ ys, const float* __restrict__ xz) {
  size_t idx = (size_t)blockIdx.x*256 + threadIdx.x;
  if (idx >= NT*ND) return;
  int d = (int)(idx & (ND-1));
  size_t t = idx >> 7;
  float z = xz[t*(2*ND) + ND + d];
  ys[idx] *= z / (1.f + expf(-z));
}

__global__ void add_inplace_kernel(float* __restrict__ a, const float* __restrict__ b, size_t n) {
  size_t i = (size_t)blockIdx.x*256 + threadIdx.x;
  if (i < n) a[i] += b[i];
}

// ---------------- layernorm over D=128 ----------------
__global__ void layernorm_kernel(const float* __restrict__ in, float* __restrict__ out,
                                 const float* __restrict__ g, const float* __restrict__ bvec) {
  size_t t = blockIdx.x;
  int d = threadIdx.x; // 128
  __shared__ float red[128];
  float x = in[t*ND + d];
  red[d] = x; __syncthreads();
  for (int s = 64; s > 0; s >>= 1) { if (d < s) red[d] += red[d+s]; __syncthreads(); }
  float mean = red[0] / (float)ND; __syncthreads();
  float c = x - mean;
  red[d] = c*c; __syncthreads();
  for (int s = 64; s > 0; s >>= 1) { if (d < s) red[d] += red[d+s]; __syncthreads(); }
  float var = red[0] / (float)ND;
  out[t*ND + d] = c * rsqrtf(var + 1e-5f) * g[d] + bvec[d];
}

// ---------------- (BV,P,D) -> (BV, D*P) for head ----------------
__global__ void transpose_kernel(const float* __restrict__ in, float* __restrict__ out) {
  size_t i = (size_t)blockIdx.x*256 + threadIdx.x;
  if (i >= (size_t)NBV*ND*NP) return;
  size_t bv = i / (ND*NP);
  int r = (int)(i % (ND*NP));
  int d = r / NP, p = r % NP;
  out[i] = in[(bv*NP + p)*ND + d];
}

// ---------------- de-normalize + transpose to (B, PRED, V) ----------------
__global__ void final_kernel(const float* __restrict__ ho, const float* __restrict__ mean,
                             const float* __restrict__ stdv, float* __restrict__ out) {
  int i = blockIdx.x*256 + threadIdx.x;
  if (i >= NBV*NPRED) return;
  int bv = i / NPRED, j = i % NPRED;
  int b = bv / NV, v = bv % NV;
  out[((size_t)b*NPRED + j)*NV + v] = ho[i]*stdv[bv] + mean[bv];
}

extern "C" void kernel_launch(void* const* d_in, const int* in_sizes, int n_in,
                              void* d_out, int out_size, void* d_ws, size_t ws_size,
                              hipStream_t stream) {
  (void)in_sizes; (void)out_size;
  if (n_in < 61) return;
  const float* P[61];
  for (int i = 0; i < 61; i++) P[i] = (const float*)d_in[i];
  // 0:x_enc 1:x_mark_enc 2:x_dec 3:x_mark_dec 4:W_emb
  // layer l at base=5+26*l: mf[0..8], mr[9..17], W_ff1 18, b_ff1 19, W_ff2 20, b_ff2 21,
  //                         ln1_g 22, ln1_b 23, ln2_g 24, ln2_b 25
  // 57:ln_f_g 58:ln_f_b 59:W_head 60:b_head
  // mamba sub-order: 0 W_in, 1 conv_w, 2 conv_b, 3 W_x, 4 W_dt, 5 b_dt, 6 A_log, 7 D, 8 W_out

  float* ws = (float*)d_ws;
  size_t off = 0;
  auto alloc = [&](size_t n) { float* p = ws + off; off += n; return p; };
  float* mean    = alloc(NBV);
  float* stdv    = alloc(NBV);
  float* buf_x   = alloc(NT*ND);       // token stream (persistent)
  float* buf_new = alloc(NT*ND);       // mamba-branch sum / x1
  float* buf_xc  = alloc(NT*ND);       // conv output / ffn output
  float* buf_ys  = alloc(NT*ND);       // scan output / y*silu(z)
  float* buf_xz  = alloc(NT*2*ND);     // in-proj / ffn hidden chunk / head input
  float* buf_dbc = alloc(NT*40);
  float* headout = alloc((size_t)NBV*NPRED);
  if (off*sizeof(float) > ws_size) return;  // insufficient workspace

  auto gemm = [&](const float* A, const float* Bw, const float* bias, float* C,
                  int M, int N, int K, int lda, int ldb, int ldc, int act, int beta) {
    dim3 grid((M + 127)/128, (N + 63)/64);
    gemm_wmma<<<grid, 256, 0, stream>>>(A, Bw, bias, C, M, N, K, lda, ldb, ldc, act, beta);
  };

  int nEl = (int)((NT*ND + 255)/256);

  auto mamba = [&](const float* x, const float* const* mp, int rev, int beta_out, float* outbuf) {
    gemm(x, mp[0], nullptr, buf_xz, (int)NT, 2*ND, ND, ND, 2*ND, 2*ND, 0, 0);
    conv_silu_kernel<<<nEl, 256, 0, stream>>>(buf_xz, mp[1], mp[2], buf_xc, rev);
    gemm(buf_xc, mp[3], nullptr, buf_dbc, (int)NT, 40, ND, ND, 40, 40, 0, 0);
    scan_kernel<<<NBV, ND, 0, stream>>>(buf_dbc, buf_xc, mp[4], mp[5], mp[6], mp[7], buf_ys, rev);
    mul_silu_kernel<<<nEl, 256, 0, stream>>>(buf_ys, buf_xz);
    gemm(buf_ys, mp[8], nullptr, outbuf, (int)NT, ND, ND, ND, ND, ND, 0, beta_out);
  };

  // ---- preprocessing ----
  stats_kernel<<<NBV, 128, 0, stream>>>(P[0], mean, stdv);
  embed_kernel<<<dim3(NBV, NP), ND, 0, stream>>>(P[0], mean, stdv, P[4], buf_x);

  // ---- encoder layers ----
  for (int l = 0; l < 2; l++) {
    int base = 5 + 26*l;
    mamba(buf_x, &P[base],     0, 0, buf_new);   // forward branch
    mamba(buf_x, &P[base + 9], 1, 1, buf_new);   // reverse branch (accumulate)
    add_inplace_kernel<<<nEl, 256, 0, stream>>>(buf_x, buf_new, NT*ND);
    layernorm_kernel<<<(unsigned)NT, ND, 0, stream>>>(buf_x, buf_new, P[base+22], P[base+23]); // x1
    // FFN 128 -> 512 -> 128, split hidden into two 256 chunks to reuse buf_xz
    for (int c = 0; c < 2; c++) {
      gemm(buf_new, P[base+18] + c*256, P[base+19] + c*256, buf_xz,
           (int)NT, 256, ND, ND, 512, 256, /*gelu*/1, 0);
      gemm(buf_xz, P[base+20] + (size_t)c*256*ND, (c == 0) ? P[base+21] : nullptr, buf_xc,
           (int)NT, ND, 256, 256, ND, ND, 0, /*beta*/c);
    }
    add_inplace_kernel<<<nEl, 256, 0, stream>>>(buf_new, buf_xc, NT*ND);
    layernorm_kernel<<<(unsigned)NT, ND, 0, stream>>>(buf_new, buf_x, P[base+24], P[base+25]);
  }

  // ---- head ----
  layernorm_kernel<<<(unsigned)NT, ND, 0, stream>>>(buf_x, buf_new, P[57], P[58]);
  transpose_kernel<<<(int)((NBV*(size_t)ND*NP + 255)/256), 256, 0, stream>>>(buf_new, buf_xz);
  gemm(buf_xz, P[59], P[60], headout, NBV, NPRED, ND*NP, ND*NP, NPRED, NPRED, 0, 0);
  final_kernel<<<(NBV*NPRED + 255)/256, 256, 0, stream>>>(headout, mean, stdv, (float*)d_out);
}
